// GNNNetwork_81406810128837
// MI455X (gfx1250) — matrix operand order repaired and checked by reading, not compiled, not run on hardware
//
#include <hip/hip_runtime.h>
#include <hip/hip_bf16.h>

// ---------------------------------------------------------------------------
// Problem constants (from the reference)
// ---------------------------------------------------------------------------
constexpr int BATCH = 1024;           // B
constexpr int IMGF  = 512;
constexpr int TXTF  = 300;
constexpr int NODES = 14;             // N
constexpr int FFD   = 2048;           // FF
constexpr int HIDD  = 128;            // HID
constexpr int CLS   = 14;             // C
constexpr int EDG   = 182;            // E
constexpr int DM    = IMGF + TXTF;    // 812
constexpr int DM3   = 3 * DM;         // 2436
constexpr int DP    = 832;            // DM padded to multiple of 32 (WMMA K)
constexpr long BN   = (long)BATCH * NODES;  // 14336 rows

// Vector types for WMMA fragments
typedef __attribute__((ext_vector_type(16))) __bf16        v16bf;
typedef __attribute__((ext_vector_type(8)))  float         v8f;
typedef __attribute__((ext_vector_type(8)))  unsigned short v8u16;

struct frag16 { v8u16 lo, hi; };  // 32 bytes == v16bf

__device__ __forceinline__ unsigned short f2bf(float f) {
    union { float f; unsigned u; } v; v.f = f;
    unsigned r = v.u + 0x7fffu + ((v.u >> 16) & 1u);   // round-to-nearest-even
    return (unsigned short)(r >> 16);
}

// Async 32-byte copy global -> LDS (two GLOBAL_LOAD_ASYNC_TO_LDS_B128).
// INST_OFFSET applies to both the LDS and global address (ISA 10. / §15.18.3),
// so offset:16 covers the second 16B chunk of a contiguous 32B segment.
// Tracked by ASYNCcnt; completion enforced with s_wait_asynccnt + barrier.
__device__ __forceinline__ void async_copy32(unsigned short* ldsDst,
                                             const unsigned short* src)
{
    unsigned dst = (unsigned)(uintptr_t)ldsDst;         // low 32 bits = LDS offset
    unsigned long long a = (unsigned long long)src;
    asm volatile("global_load_async_to_lds_b128 %0, %1, off\n\t"
                 "global_load_async_to_lds_b128 %0, %1, off offset:16"
                 :: "v"(dst), "v"(a) : "memory");
}
__device__ __forceinline__ void wait_async0()
{
    asm volatile("s_wait_asynccnt 0x0" ::: "memory");
}

// ---------------------------------------------------------------------------
// Generic bf16 WMMA GEMM:  C[b] = act( A[b] (MxK) * Bt[b]^T (NoutxK) * alpha + bias )
// Bt stored row-major (Nout, Kp): row = output column, K contiguous -> ideal
// for the CDNA5 B-operand (one column per lane, K along VGPRs).
//
// Block: 256 threads = 8 waves. Block tile 128x64, K-step 32, LDS double
// buffered with async global->LDS copies (ASYNCcnt) overlapping the WMMA
// compute of the live buffer. All four B fragments are loaded into distinct
// registers before any WMMA issues, so the K-step takes ONE s_wait_dscnt and
// then fires 4 back-to-back v_wmma (XDL co-execution friendly).
// M must be a multiple of 128 and Kp a multiple of 32 (all buffers padded);
// Nout guarded (EXEC-masked async loads; stale LDS columns never stored).
// ---------------------------------------------------------------------------
__global__ __launch_bounds__(256)
void gemm_wmma_bf16(const unsigned short* __restrict__ A,  long lda, long sA,
                    const unsigned short* __restrict__ Bt, long ldb, long sB,
                    const float* __restrict__ bias,
                    float* __restrict__ Cf, unsigned short* __restrict__ Cb,
                    long ldc, long sC,
                    int M, int Nout, int Kp, float alpha, int act)
{
    __shared__ unsigned short As[2][128][40];   // +8 skew: 16B-aligned, conflict-free
    __shared__ unsigned short Bs[2][64][40];

    const int tid   = threadIdx.x;
    const int wave  = tid >> 5;
    const int lane  = tid & 31;
    const int batch = blockIdx.z;
    const long rowBase = (long)blockIdx.y * 128;
    const long colBase = (long)blockIdx.x * 64;

    const unsigned short* Ab = A  + (long)batch * sA;
    const unsigned short* Bb = Bt + (long)batch * sB;

    v8f zero8 = {0.f,0.f,0.f,0.f,0.f,0.f,0.f,0.f};
    v8f acc[4]; acc[0] = zero8; acc[1] = zero8; acc[2] = zero8; acc[3] = zero8;

    // cooperative loaders: each thread owns one 32B segment
    const int aRow = tid >> 1;            // 0..127
    const int aSeg = (tid & 1) * 16;
    const int bRow = (tid & 127) >> 1;    // 0..63
    const int bSeg = (tid & 1) * 16;
    const long bCol = colBase + bRow;

    auto issue_tile = [&](int ks, int buf) {
        const long k0 = (long)ks << 5;
        async_copy32(&As[buf][aRow][aSeg], Ab + (rowBase + aRow) * lda + k0 + aSeg);
        if (tid < 128 && bCol < Nout)
            async_copy32(&Bs[buf][bRow][bSeg], Bb + bCol * ldb + k0 + bSeg);
    };

    const int ksteps = Kp >> 5;
    issue_tile(0, 0);

    for (int ks = 0; ks < ksteps; ++ks) {
        wait_async0();          // own async copies done ...
        __syncthreads();        // ... then everyone's (barrier implies dscnt drain)

        if (ks + 1 < ksteps) issue_tile(ks + 1, (ks + 1) & 1);
        if (ks + 2 < ksteps) {  // warm L2 for the tile after next
            const long k2 = (long)(ks + 2) << 5;
            __builtin_prefetch(Ab + (rowBase + aRow) * lda + k2 + aSeg, 0, 3);
            if (tid < 128 && bCol < Nout)
                __builtin_prefetch(Bb + bCol * ldb + k2 + bSeg, 0, 3);
        }

        const int buf = ks & 1;
        // A fragment, 16-bit 16x32 layout: lanes 0-15 row M=lane hold K 0..7 /
        // 16..23; lanes 16-31 hold K 8..15 / 24..31.
        const int r  = lane & 15;
        const int hb = lane >> 4;
        frag16 af;
        af.lo = *(const v8u16*)&As[buf][wave * 16 + r][hb * 8];
        af.hi = *(const v8u16*)&As[buf][wave * 16 + r][16 + hb * 8];
        v16bf a = __builtin_bit_cast(v16bf, af);

        // Load ALL four B fragments first (one ds_load clause, distinct regs),
        // then issue the 4 WMMAs back-to-back behind a single dscnt wait.
        frag16 bf4[4];
        #pragma unroll
        for (int j = 0; j < 4; ++j) {
            bf4[j].lo = *(const v8u16*)&Bs[buf][j * 16 + r][hb * 8];
            bf4[j].hi = *(const v8u16*)&Bs[buf][j * 16 + r][16 + hb * 8];
        }
        #pragma unroll
        for (int j = 0; j < 4; ++j) {
            v16bf bv = __builtin_bit_cast(v16bf, bf4[j]);
            acc[j] = __builtin_amdgcn_wmma_f32_16x16x32_bf16(
                false, a, false, bv, (short)0, acc[j], false, false);
        }
        __syncthreads();
    }

    // Epilogue. C/D layout: lane l, VGPR i -> M = i + (l>=16)*8, N = l&15.
    const int r  = lane & 15;
    const int hb = lane >> 4;
    #pragma unroll
    for (int j = 0; j < 4; ++j) {
        const long col = colBase + j * 16 + r;
        if (col >= Nout) continue;
        const float bv = bias ? bias[col] : 0.0f;
        #pragma unroll
        for (int i = 0; i < 8; ++i) {
            const long row = rowBase + wave * 16 + hb * 8 + i;
            float v = acc[j][i] * alpha + bv;
            if (act == 1) v = fmaxf(v, 0.0f);
            const long off = (long)batch * sC + row * ldc + col;
            if (Cf) Cf[off] = v;
            if (Cb) Cb[off] = f2bf(v);
        }
    }
}

// ---------------------------------------------------------------------------
// Weight pack: f32 -> bf16, optional transpose, zero K-padding.
// dst[o*Kp + k] = (k<K) ? (trans ? src[k*O+o] : src[o*K+k]) : 0
// ---------------------------------------------------------------------------
__global__ void pack_weight(const float* __restrict__ src, unsigned short* __restrict__ dst,
                            int O, int K, int Kp, int trans)
{
    long t = (long)blockIdx.x * blockDim.x + threadIdx.x;
    if (t >= (long)O * Kp) return;
    int o = (int)(t / Kp), k = (int)(t % Kp);
    float v = 0.0f;
    if (k < K) v = trans ? src[(long)k * O + o] : src[(long)o * K + k];
    dst[t] = f2bf(v);
}

// node = concat(img broadcast over N, word_embeds broadcast over B)
__global__ void build_node(const float* __restrict__ img, const float* __restrict__ wemb,
                           float* __restrict__ nodeF, unsigned short* __restrict__ nodeB)
{
    long t = (long)blockIdx.x * blockDim.x + threadIdx.x;
    if (t >= BN * DP) return;
    long m = t / DP; int dp = (int)(t % DP);
    int b = (int)(m / NODES), n = (int)(m % NODES);
    float v = 0.f;
    if (dp < IMGF)      v = img[(long)b * IMGF + dp];
    else if (dp < DM)   v = wemb[(long)n * TXTF + (dp - IMGF)];
    if (dp < DM) nodeF[m * DM + dp] = v;
    nodeB[t] = f2bf(v);
}

// Repack qkv (row = b*N+n, 2436 cols) into per-node operands:
//  qP,kP : (n, b, DP) bf16 row-major (zero padded)   -> A / Bt of scores GEMM
//  vT    : (n, DP, b) bf16, vT[n][d][b] = v[b*N+n][d] -> Bt of attn*V GEMM
__global__ void pack_qkv(const unsigned short* __restrict__ qkv,
                         unsigned short* __restrict__ qP,
                         unsigned short* __restrict__ kP,
                         unsigned short* __restrict__ vT)
{
    long t = (long)blockIdx.x * blockDim.x + threadIdx.x;
    if (t >= (long)NODES * BATCH * DP) return;
    int dp = (int)(t % DP);
    long nb = t / DP;
    int b = (int)(nb % BATCH), n = (int)(nb / BATCH);
    long m = (long)b * NODES + n;
    unsigned short qv = 0, kv = 0;
    if (dp < DM) {
        qv = qkv[m * DM3 + dp];
        kv = qkv[m * DM3 + DM + dp];
        vT[((long)n * DP + dp) * BATCH + b] = qkv[m * DM3 + 2 * DM + dp];
    }
    qP[t] = qv;
    kP[t] = kv;
}

// Row softmax over width 1024 (rows = n*B + l), f32 in -> bf16 out
__global__ __launch_bounds__(256)
void softmax_rows(const float* __restrict__ S, unsigned short* __restrict__ P)
{
    __shared__ float red[256];
    const long row = blockIdx.x;
    const float* s = S + row * BATCH;
    float v[4]; float m = -1e30f;
    #pragma unroll
    for (int i = 0; i < 4; ++i) { v[i] = s[threadIdx.x + i * 256]; m = fmaxf(m, v[i]); }
    red[threadIdx.x] = m; __syncthreads();
    for (int st = 128; st > 0; st >>= 1) {
        if (threadIdx.x < st) red[threadIdx.x] = fmaxf(red[threadIdx.x], red[threadIdx.x + st]);
        __syncthreads();
    }
    m = red[0]; __syncthreads();
    float sum = 0.f;
    #pragma unroll
    for (int i = 0; i < 4; ++i) { v[i] = __expf(v[i] - m); sum += v[i]; }
    red[threadIdx.x] = sum; __syncthreads();
    for (int st = 128; st > 0; st >>= 1) {
        if (threadIdx.x < st) red[threadIdx.x] += red[threadIdx.x + st];
        __syncthreads();
    }
    const float inv = 1.0f / red[0];
    unsigned short* p = P + row * BATCH;
    #pragma unroll
    for (int i = 0; i < 4; ++i) p[threadIdx.x + i * 256] = f2bf(v[i] * inv);
}

// Zero the K-pad columns of a (BN, DP) bf16 buffer (cols DM..DP-1)
__global__ void zero_pad_cols(unsigned short* __restrict__ buf)
{
    long t = (long)blockIdx.x * blockDim.x + threadIdx.x;
    if (t >= BN * (DP - DM)) return;
    long m = t / (DP - DM); int j = (int)(t % (DP - DM));
    buf[m * DP + DM + j] = 0;
}

// x = LayerNorm(X + Y) * g + be ; writes f32 (optional) + padded bf16
__global__ __launch_bounds__(256)
void residual_layernorm(const float* __restrict__ X, const float* __restrict__ Y,
                        const float* __restrict__ g, const float* __restrict__ be,
                        float* __restrict__ outF, unsigned short* __restrict__ outB)
{
    __shared__ float red[256];
    const long row = blockIdx.x;
    const float* x = X + row * DM;
    const float* y = Y + row * DM;
    float v[4]; float sum = 0.f;
    #pragma unroll
    for (int i = 0; i < 4; ++i) {
        int d = threadIdx.x + i * 256;
        v[i] = (d < DM) ? (x[d] + y[d]) : 0.f;
        sum += v[i];
    }
    red[threadIdx.x] = sum; __syncthreads();
    for (int st = 128; st > 0; st >>= 1) {
        if (threadIdx.x < st) red[threadIdx.x] += red[threadIdx.x + st];
        __syncthreads();
    }
    const float mu = red[0] / (float)DM;
    __syncthreads();
    float sq = 0.f;
    #pragma unroll
    for (int i = 0; i < 4; ++i) {
        int d = threadIdx.x + i * 256;
        float tt = (d < DM) ? (v[i] - mu) : 0.f;
        sq += tt * tt;
    }
    red[threadIdx.x] = sq; __syncthreads();
    for (int st = 128; st > 0; st >>= 1) {
        if (threadIdx.x < st) red[threadIdx.x] += red[threadIdx.x + st];
        __syncthreads();
    }
    const float rstd = rsqrtf(red[0] / (float)DM + 1e-5f);
    #pragma unroll
    for (int i = 0; i < 4; ++i) {
        int d = threadIdx.x + i * 256;
        if (d < DM) {
            float o = (v[i] - mu) * rstd * g[d] + be[d];
            if (outF) outF[row * DM + d] = o;
            outB[row * DP + d] = f2bf(o);
        }
    }
    if (threadIdx.x < DP - DM) outB[row * DP + DM + threadIdx.x] = 0;
}

// Build dense normalized adjacency (with self loops): Ahat[dst][src] += dinv*dinv
__global__ __launch_bounds__(64)
void build_ahat(const int* __restrict__ ei, float* __restrict__ AhatG)
{
    __shared__ float deg[NODES];
    __shared__ float dinv[NODES];
    __shared__ float Ah[NODES * NODES];
    const int tid = threadIdx.x;
    if (tid < NODES) deg[tid] = 0.f;
    for (int i = tid; i < NODES * NODES; i += 64) Ah[i] = 0.f;
    __syncthreads();
    for (int e = tid; e < EDG + NODES; e += 64) {
        int dst = (e < EDG) ? ei[EDG + e] : (e - EDG);
        atomicAdd(&deg[dst], 1.0f);
    }
    __syncthreads();
    if (tid < NODES) dinv[tid] = (deg[tid] > 0.f) ? rsqrtf(deg[tid]) : 0.f;
    __syncthreads();
    for (int e = tid; e < EDG + NODES; e += 64) {
        int src = (e < EDG) ? ei[e] : (e - EDG);
        int dst = (e < EDG) ? ei[EDG + e] : (e - EDG);
        atomicAdd(&Ah[dst * NODES + src], dinv[src] * dinv[dst]);
    }
    __syncthreads();
    for (int i = tid; i < NODES * NODES; i += 64) AhatG[i] = Ah[i];
}

// GCN1 aggregation: out[b,j,c] = relu( sum_i Ahat[j,i]*h[b,i,c] + bias[c] ) -> bf16
__global__ __launch_bounds__(256)
void gcn_aggregate_relu(const float* __restrict__ h, const float* __restrict__ Ahat,
                        const float* __restrict__ bias, unsigned short* __restrict__ outB)
{
    __shared__ float Ah[NODES * NODES];
    if (threadIdx.x < NODES * NODES) Ah[threadIdx.x] = Ahat[threadIdx.x];
    __syncthreads();
    long t = (long)blockIdx.x * 256 + threadIdx.x;
    if (t >= BN * HIDD) return;
    int c = (int)(t % HIDD);
    long bj = t / HIDD; int j = (int)(bj % NODES); long b = bj / NODES;
    float s = bias[c];
    #pragma unroll
    for (int i = 0; i < NODES; ++i)
        s += Ah[j * NODES + i] * h[(b * NODES + i) * HIDD + c];
    outB[t] = f2bf(fmaxf(s, 0.f));
}

// Fused GCN2-aggregate + node-mean + relu + final linear:
// pooled[c] = relu( sum_i (mean_j Ahat[j,i]) * h2[b,i,c] + b2[c] )
// out[b,cc] = pooled . linW[:,cc] + linB[cc]
__global__ __launch_bounds__(128)
void gcn2_pool_linear(const float* __restrict__ h2, const float* __restrict__ Ahat,
                      const float* __restrict__ b2, const float* __restrict__ linW,
                      const float* __restrict__ linB, float* __restrict__ out)
{
    __shared__ float Ah[NODES * NODES];
    __shared__ float sc[NODES];
    __shared__ float pooled[HIDD];
    const int tid = threadIdx.x;
    const long b = blockIdx.x;
    if (tid < NODES * NODES) Ah[tid] = Ahat[tid];
    __syncthreads();
    if (tid < NODES) {
        float s = 0.f;
        for (int j = 0; j < NODES; ++j) s += Ah[j * NODES + tid];
        sc[tid] = s / (float)NODES;
    }
    __syncthreads();
    {
        const int c = tid;  // blockDim == HIDD == 128
        float v = b2[c];
        #pragma unroll
        for (int i = 0; i < NODES; ++i)
            v += sc[i] * h2[(b * NODES + i) * HIDD + c];
        pooled[c] = fmaxf(v, 0.f);
    }
    __syncthreads();
    if (tid < CLS) {
        float o = linB[tid];
        for (int c = 0; c < HIDD; ++c) o += pooled[c] * linW[c * CLS + tid];
        out[b * CLS + tid] = o;
    }
}

// ---------------------------------------------------------------------------
extern "C" void kernel_launch(void* const* d_in, const int* in_sizes, int n_in,
                              void* d_out, int out_size, void* d_ws, size_t ws_size,
                              hipStream_t stream)
{
    const float* img  = (const float*)d_in[0];
    const float* wemb = (const float*)d_in[1];
    const int*   ei   = (const int*)d_in[2];
    const float* in_w = (const float*)d_in[3];
    const float* in_b = (const float*)d_in[4];
    const float* ow   = (const float*)d_in[5];
    const float* ob   = (const float*)d_in[6];
    const float* ln1g = (const float*)d_in[7];
    const float* ln1b = (const float*)d_in[8];
    const float* ln2g = (const float*)d_in[9];
    const float* ln2b = (const float*)d_in[10];
    const float* w1   = (const float*)d_in[11];
    const float* b1   = (const float*)d_in[12];
    const float* w2   = (const float*)d_in[13];
    const float* b2   = (const float*)d_in[14];
    const float* g1w  = (const float*)d_in[15];
    const float* g1b  = (const float*)d_in[16];
    const float* g2w  = (const float*)d_in[17];
    const float* g2b  = (const float*)d_in[18];
    const float* lw   = (const float*)d_in[19];
    const float* lb   = (const float*)d_in[20];
    float* out = (float*)d_out;

    // ---- workspace carve-up (~620 MB total) ----
    char* wsp = (char*)d_ws;
    size_t off = 0;
    auto alloc = [&](size_t elems, size_t esz) -> void* {
        void* p = wsp + off;
        off = (off + elems * esz + 255) & ~(size_t)255;
        return p;
    };
    float*          nodeF = (float*)alloc(BN * DM, 4);
    unsigned short* nodeB = (unsigned short*)alloc(BN * DP, 2);
    unsigned short* Wqkv  = (unsigned short*)alloc((size_t)DM3 * DP, 2);
    unsigned short* Woutp = (unsigned short*)alloc((size_t)DM * DP, 2);
    unsigned short* W1t   = (unsigned short*)alloc((size_t)FFD * DP, 2);
    unsigned short* W2t   = (unsigned short*)alloc((size_t)DM * FFD, 2);
    unsigned short* G1t   = (unsigned short*)alloc((size_t)HIDD * DP, 2);
    unsigned short* G2t   = (unsigned short*)alloc((size_t)HIDD * HIDD, 2);
    unsigned short* qkvB  = (unsigned short*)alloc(BN * DM3, 2);
    unsigned short* qP    = (unsigned short*)alloc((size_t)NODES * BATCH * DP, 2);
    unsigned short* kP    = (unsigned short*)alloc((size_t)NODES * BATCH * DP, 2);
    unsigned short* vT    = (unsigned short*)alloc((size_t)NODES * DP * BATCH, 2);
    float*          scoresF = (float*)alloc((size_t)NODES * BATCH * BATCH, 4);
    unsigned short* attnB = (unsigned short*)alloc((size_t)NODES * BATCH * BATCH, 2);
    unsigned short* aoB   = (unsigned short*)alloc(BN * DP, 2);
    float*          projF = (float*)alloc(BN * DM, 4);
    float*          xF    = (float*)alloc(BN * DM, 4);
    unsigned short* xB    = (unsigned short*)alloc(BN * DP, 2);
    unsigned short* hB    = (unsigned short*)alloc(BN * FFD, 2);
    float*          ffF   = (float*)alloc(BN * DM, 4);
    unsigned short* x2B   = (unsigned short*)alloc(BN * DP, 2);
    float*          h1F   = (float*)alloc(BN * HIDD, 4);
    unsigned short* a1B   = (unsigned short*)alloc(BN * HIDD, 2);
    float*          h2F   = (float*)alloc(BN * HIDD, 4);
    float*          Ahat  = (float*)alloc(NODES * NODES, 4);

    auto gemm = [&](const unsigned short* A, long lda, long sA,
                    const unsigned short* Bt, long ldb, long sB,
                    const float* bias, float* Cf, unsigned short* Cb,
                    long ldc, long sC, int M, int Nout, int Kp,
                    float alpha, int act, int batch) {
        dim3 grid((Nout + 63) / 64, M / 128, batch);
        gemm_wmma_bf16<<<grid, dim3(256), 0, stream>>>(
            A, lda, sA, Bt, ldb, sB, bias, Cf, Cb, ldc, sC, M, Nout, Kp, alpha, act);
    };
    auto gridFor = [](long total) { return (unsigned)((total + 255) / 256); };

    // ---- weight packs (f32 -> padded bf16, transposed where K must be contiguous) ----
    pack_weight<<<gridFor((long)DM3 * DP), 256, 0, stream>>>(in_w, Wqkv, DM3, DM, DP, 0);
    pack_weight<<<gridFor((long)DM  * DP), 256, 0, stream>>>(ow,   Woutp, DM, DM, DP, 0);
    pack_weight<<<gridFor((long)FFD * DP), 256, 0, stream>>>(w1,   W1t, FFD, DM, DP, 1);
    pack_weight<<<gridFor((long)DM * FFD), 256, 0, stream>>>(w2,   W2t, DM, FFD, FFD, 1);
    pack_weight<<<gridFor((long)HIDD * DP), 256, 0, stream>>>(g1w, G1t, HIDD, DM, DP, 1);
    pack_weight<<<gridFor((long)HIDD * HIDD), 256, 0, stream>>>(g2w, G2t, HIDD, HIDD, HIDD, 1);
    build_ahat<<<1, 64, 0, stream>>>(ei, Ahat);

    // ---- node features ----
    build_node<<<gridFor(BN * DP), 256, 0, stream>>>(img, wemb, nodeF, nodeB);

    // ---- QKV projection: (14336,832)x(2436,832)^T -> bf16 qkv ----
    gemm(nodeB, DP, 0, Wqkv, DP, 0, in_b, nullptr, qkvB, DM3, 0,
         (int)BN, DM3, DP, 1.0f, 0, 1);
    pack_qkv<<<gridFor((long)NODES * BATCH * DP), 256, 0, stream>>>(qkvB, qP, kP, vT);

    // ---- attention scores per node: (1024,832)x(1024,832)^T * 1/sqrt(D) ----
    const float scale = 1.0f / sqrtf((float)DM);
    gemm(qP, DP, (long)BATCH * DP, kP, DP, (long)BATCH * DP, nullptr,
         scoresF, nullptr, BATCH, (long)BATCH * BATCH,
         BATCH, BATCH, DP, scale, 0, NODES);
    softmax_rows<<<NODES * BATCH, 256, 0, stream>>>(scoresF, attnB);

    // ---- attn @ V -> ao laid out directly as (b*N+n, DP) bf16 ----
    zero_pad_cols<<<gridFor(BN * (DP - DM)), 256, 0, stream>>>(aoB);
    gemm(attnB, BATCH, (long)BATCH * BATCH, vT, BATCH, (long)DP * BATCH, nullptr,
         nullptr, aoB, (long)NODES * DP, DP,
         BATCH, DM, BATCH, 1.0f, 0, NODES);

    // ---- output projection + residual LN1 ----
    gemm(aoB, DP, 0, Woutp, DP, 0, ob, projF, nullptr, DM, 0,
         (int)BN, DM, DP, 1.0f, 0, 1);
    residual_layernorm<<<(unsigned)BN, 256, 0, stream>>>(nodeF, projF, ln1g, ln1b, xF, xB);

    // ---- FFN ----
    gemm(xB, DP, 0, W1t, DP, 0, b1, nullptr, hB, FFD, 0,
         (int)BN, FFD, DP, 1.0f, 1, 1);                           // relu -> bf16
    gemm(hB, FFD, 0, W2t, FFD, 0, b2, ffF, nullptr, DM, 0,
         (int)BN, DM, FFD, 1.0f, 0, 1);
    residual_layernorm<<<(unsigned)BN, 256, 0, stream>>>(xF, ffF, ln2g, ln2b, nullptr, x2B);

    // ---- GCN layers over dense 14x14 normalized adjacency ----
    gemm(x2B, DP, 0, G1t, DP, 0, nullptr, h1F, nullptr, HIDD, 0,
         (int)BN, HIDD, DP, 1.0f, 0, 1);
    gcn_aggregate_relu<<<gridFor(BN * HIDD), 256, 0, stream>>>(h1F, Ahat, g1b, a1B);
    gemm(a1B, HIDD, 0, G2t, HIDD, 0, nullptr, h2F, nullptr, HIDD, 0,
         (int)BN, HIDD, HIDD, 1.0f, 0, 1);
    gcn2_pool_linear<<<BATCH, HIDD, 0, stream>>>(h2F, Ahat, g2b, lw, lb, out);
}
